// Block_60249801228390
// MI455X (gfx1250) — compile-verified, matrix operand order checked by reading, not run on hardware
//
#include <hip/hip_runtime.h>
#include <math.h>

#define NJ    17
#define DIMC  512
#define HEADS 8
#define HDC   64
#define MIDC  256
#define BATCH 2048
#define MROWS (BATCH * NJ)   // 34816 = 1088 * 32

typedef __attribute__((ext_vector_type(16))) __bf16 v16bf;
typedef __attribute__((ext_vector_type(8)))  float  v8f;

// Human3.6M skeleton: parent of each joint (index 0 unused sentinel)
__constant__ int cPAR[NJ] = {0,0,1,2,0,4,5,0,7,8,9,8,11,12,8,14,15};
// top-level ancestor (direct child of root) per joint
__constant__ int cANC[NJ] = {0,1,1,1,4,4,4,7,7,7,7,7,7,7,7,7,7};

// ---------------------------------------------------------------------------
// bf16-WMMA GEMM:  C[M,Nd] = act(A[M,Kd] @ W[Nd,Kd]^T + bias) (+ res)
// Wave tile 32x64: 2 A frags x 4 B frags -> 8 WMMAs per 32-deep k-step.
// Loads for the whole k-step are issued before any conversion so the
// compiler can batch them (loadcnt > 0) instead of wait-after-each-load.
// act: 0=none, 1=LeakyReLU(0.01), 2=GELU(erf)
// ---------------------------------------------------------------------------
__global__ __launch_bounds__(128)
void gemm_wmma_kernel(const float* __restrict__ A, const float* __restrict__ W,
                      const float* __restrict__ bias, const float* __restrict__ res,
                      float* __restrict__ C, int Kd, int Nd, int act)
{
    const int lane = threadIdx.x & 31;
    const int wave = threadIdx.x >> 5;
    const int row0 = blockIdx.x << 5;                 // 32 rows per block
    const int col0 = (blockIdx.y * 4 + wave) << 6;    // 64 cols per wave
    const int lrow = lane & 15;
    const int khi  = (lane >> 4) << 3;   // lanes 0-15: K 0..7/16..23, lanes 16-31: 8..15/24..31

    v8f acc[8];
#pragma unroll
    for (int i = 0; i < 8; ++i) acc[i] = v8f{};

    const float* a0 = A + (size_t)(row0 + lrow) * Kd + khi;
    const float* a1 = A + (size_t)(row0 + 16 + lrow) * Kd + khi;
    const float* w0 = W + (size_t)(col0 + lrow) * Kd + khi;

    for (int k0 = 0; k0 < Kd; k0 += 32) {
        // ---- phase 1: issue all global loads for this k-step ----
        float4 ra[2][4];
        float4 rb[4][4];
        {
            const float4* p0 = (const float4*)(a0 + k0);
            const float4* p1 = (const float4*)(a1 + k0);
            ra[0][0] = p0[0]; ra[0][1] = p0[1]; ra[0][2] = p0[4]; ra[0][3] = p0[5];
            ra[1][0] = p1[0]; ra[1][1] = p1[1]; ra[1][2] = p1[4]; ra[1][3] = p1[5];
        }
#pragma unroll
        for (int t = 0; t < 4; ++t) {
            const float4* pb = (const float4*)(w0 + (size_t)(t * 16) * Kd + k0);
            rb[t][0] = pb[0]; rb[t][1] = pb[1]; rb[t][2] = pb[4]; rb[t][3] = pb[5];
        }
        // ---- phase 2: convert everything to bf16 fragments ----
        v16bf fga[2], fgb[4];
#pragma unroll
        for (int m = 0; m < 2; ++m) {
            const float* f = (const float*)&ra[m][0];
#pragma unroll
            for (int i = 0; i < 16; ++i) fga[m][i] = (__bf16)f[i];
        }
#pragma unroll
        for (int t = 0; t < 4; ++t) {
            const float* f = (const float*)&rb[t][0];
#pragma unroll
            for (int i = 0; i < 16; ++i) fgb[t][i] = (__bf16)f[i];
        }
        // ---- phase 3: 8 back-to-back WMMAs ----
#pragma unroll
        for (int t = 0; t < 4; ++t) {
            acc[t] = __builtin_amdgcn_wmma_f32_16x16x32_bf16(
                false, fga[0], false, fgb[t], (short)0, acc[t], false, false);
            acc[4 + t] = __builtin_amdgcn_wmma_f32_16x16x32_bf16(
                false, fga[1], false, fgb[t], (short)0, acc[4 + t], false, false);
        }
    }

    // C/D layout: VGPR r -> M = r (+8 for upper lanes), N = lane&15
    const int rbase = ((lane >> 4) << 3);
    const int ocol  = lane & 15;
#pragma unroll
    for (int m = 0; m < 2; ++m) {
        const int orow0 = row0 + m * 16 + rbase;
#pragma unroll
        for (int t = 0; t < 4; ++t) {
            const int col = col0 + t * 16 + ocol;
            const float bsv = bias ? bias[col] : 0.0f;
#pragma unroll
            for (int r = 0; r < 8; ++r) {
                float v = acc[m * 4 + t][r] + bsv;
                if (act == 1)      v = (v >= 0.0f) ? v : 0.01f * v;
                else if (act == 2) v = 0.5f * v * (1.0f + erff(v * 0.70710678118654752f));
                const size_t o = (size_t)(orow0 + r) * Nd + col;
                if (res) v += res[o];
                C[o] = v;
            }
        }
    }
}

// ---------------------------------------------------------------------------
// prep: Ahat = D^-1/2 (A+I) D^-1/2 (17x17) and rd = (1+e)/(1+exp(1-dist))
// ---------------------------------------------------------------------------
__global__ void prep_kernel(const float* __restrict__ ew, const float* __restrict__ rdist,
                            const int* __restrict__ ei, float* __restrict__ Ahat,
                            float* __restrict__ rd)
{
    __shared__ float Am[NJ * NJ];
    __shared__ float dn[NJ];
    const int tid = threadIdx.x;
    for (int i = tid; i < NJ * NJ; i += blockDim.x) Am[i] = 0.0f;
    __syncthreads();
    if (tid == 0) {
        for (int e = 0; e < 32; ++e) Am[ei[e] * NJ + ei[32 + e]] += ew[e];
        for (int i = 0; i < NJ; ++i) Am[i * NJ + i] += 1.0f;
    }
    __syncthreads();
    if (tid < NJ) {
        float s = 0.0f;
        for (int w = 0; w < NJ; ++w) s += Am[tid * NJ + w];
        dn[tid] = rsqrtf(s);
    }
    __syncthreads();
    for (int i = tid; i < NJ * NJ; i += blockDim.x) {
        const int v = i / NJ, w = i % NJ;
        Ahat[i] = dn[v] * Am[i] * dn[w];
        rd[i]   = (1.0f + 2.7182818284590452f) / (1.0f + expf(1.0f - rdist[i]));
    }
}

__global__ void zero_kernel(float* __restrict__ p, int n)
{
    const int i = blockIdx.x * blockDim.x + threadIdx.x;
    if (i < n) p[i] = 0.0f;
}

// ---------------------------------------------------------------------------
// Joint_GCN mix: x1[b,w,c] = x[b,w,c] + sum_{k,v} H[b,v,k,c] * adj_k[k,v,w]
// grid (B, 2): each block handles a 256-channel half.
// ---------------------------------------------------------------------------
__global__ __launch_bounds__(256)
void jgcn_mix_kernel(const float* __restrict__ x, const float* __restrict__ H,
                     const float* __restrict__ adjk, float* __restrict__ xout)
{
    __shared__ float Hs[NJ * 3 * 256];   // [v][k][cl]
    __shared__ float As[3 * NJ * NJ];
    const int b = blockIdx.x, c0 = blockIdx.y * 256, tid = threadIdx.x;
    for (int idx = tid; idx < NJ * 3 * 256; idx += 256) {
        const int v = idx / 768, r = idx % 768, k = r / 256, cl = r % 256;
        Hs[idx] = H[((size_t)b * NJ + v) * (3 * DIMC) + k * DIMC + c0 + cl];
    }
    for (int idx = tid; idx < 3 * NJ * NJ; idx += 256) As[idx] = adjk[idx];
    __syncthreads();
    for (int idx = tid; idx < NJ * 256; idx += 256) {
        const int w = idx / 256, cl = idx % 256;
        float s = 0.0f;
        for (int k = 0; k < 3; ++k)
            for (int v = 0; v < NJ; ++v)
                s += Hs[v * 768 + k * 256 + cl] * As[k * (NJ * NJ) + v * NJ + w];
        const size_t o = ((size_t)b * NJ + w) * DIMC + c0 + cl;
        xout[o] = x[o] + s;
    }
}

// ---------------------------------------------------------------------------
// Bone mix: vec = vector + Ahat@T1 + gcn1_b + 0.5 * (T3 mixed by dir_adj)
// ---------------------------------------------------------------------------
__global__ __launch_bounds__(256)
void bone_mix_kernel(const float* __restrict__ vin, const float* __restrict__ T1,
                     const float* __restrict__ T3, const float* __restrict__ Ahat,
                     const float* __restrict__ dadj, const float* __restrict__ gb,
                     float* __restrict__ vout)
{
    __shared__ float T1s[NJ * 256], T3s[NJ * 256], Ah[NJ * NJ], Da[NJ * NJ];
    const int b = blockIdx.x, c0 = blockIdx.y * 256, tid = threadIdx.x;
    for (int idx = tid; idx < NJ * 256; idx += 256) {
        const int v = idx / 256, cl = idx % 256;
        const size_t o = ((size_t)b * NJ + v) * DIMC + c0 + cl;
        T1s[idx] = T1[o];
        T3s[idx] = T3[o];
    }
    for (int idx = tid; idx < NJ * NJ; idx += 256) { Ah[idx] = Ahat[idx]; Da[idx] = dadj[idx]; }
    __syncthreads();
    for (int idx = tid; idx < NJ * 256; idx += 256) {
        const int j = idx / 256, cl = idx % 256;
        float g = 0.0f, d = 0.0f;
        for (int w = 0; w < NJ; ++w) g += Ah[j * NJ + w] * T1s[w * 256 + cl];
        for (int v = 0; v < NJ; ++v) d += T3s[v * 256 + cl] * Da[v * NJ + j];
        const size_t o = ((size_t)b * NJ + j) * DIMC + c0 + cl;
        vout[o] = vin[o] + g + gb[c0 + cl] + 0.5f * d;
    }
}

// ---------------------------------------------------------------------------
// LayerNorm: one wave per row (C=512), 8 rows per 256-thread block
// ---------------------------------------------------------------------------
__global__ __launch_bounds__(256)
void ln_kernel(const float* __restrict__ x, const float* __restrict__ g,
               const float* __restrict__ bb, float* __restrict__ out)
{
    const int row  = blockIdx.x * 8 + (threadIdx.x >> 5);
    const int lane = threadIdx.x & 31;
    const float* xr = x + (size_t)row * DIMC;
    float s = 0.0f, s2 = 0.0f;
    for (int c = lane; c < DIMC; c += 32) { const float v = xr[c]; s += v; s2 += v * v; }
#pragma unroll
    for (int o = 16; o > 0; o >>= 1) { s += __shfl_xor(s, o, 32); s2 += __shfl_xor(s2, o, 32); }
    const float m   = s * (1.0f / DIMC);
    const float var = s2 * (1.0f / DIMC) - m * m;
    const float r   = rsqrtf(var + 1e-5f);
    float* orow = out + (size_t)row * DIMC;
    for (int c = lane; c < DIMC; c += 32) orow[c] = (xr[c] - m) * r * g[c] + bb[c];
}

// ---------------------------------------------------------------------------
// CoAttention: per (batch, 4-head group): softmax(q k^T / sqrt(hd)) @ v,
// accumulate scores[b,n] += mean_h sum_m attn
// ---------------------------------------------------------------------------
__global__ __launch_bounds__(256)
void co_attn_kernel(const float* __restrict__ qkv, float* __restrict__ ao,
                    float* __restrict__ scores)
{
    __shared__ float qs[4 * NJ * HDC], ks[4 * NJ * HDC], vs[4 * NJ * HDC];
    __shared__ float S[4 * NJ * NJ];
    const int b = blockIdx.x, hg = blockIdx.y, tid = threadIdx.x;
    for (int idx = tid; idx < 4 * NJ * HDC; idx += 256) {
        const int hh = idx / (NJ * HDC), r = idx % (NJ * HDC), n = r / HDC, d = r % HDC;
        const int h = hg * 4 + hh;
        const size_t base = ((size_t)b * NJ + n) * (3 * DIMC);
        qs[idx] = qkv[base + h * HDC + d];
        ks[idx] = qkv[base + DIMC + h * HDC + d];
        vs[idx] = qkv[base + 2 * DIMC + h * HDC + d];
    }
    __syncthreads();
    for (int idx = tid; idx < 4 * NJ * NJ; idx += 256) {
        const int hh = idx / (NJ * NJ), r = idx % (NJ * NJ), n = r / NJ, m = r % NJ;
        const float* qp = &qs[(hh * NJ + n) * HDC];
        const float* kp = &ks[(hh * NJ + m) * HDC];
        float acc = 0.0f;
        for (int d = 0; d < HDC; ++d) acc += qp[d] * kp[d];
        S[idx] = acc * 0.125f;   // HD^-0.5
    }
    __syncthreads();
    for (int idx = tid; idx < 4 * NJ; idx += 256) {
        float* row = &S[idx * NJ];
        float mx = row[0];
        for (int m = 1; m < NJ; ++m) mx = fmaxf(mx, row[m]);
        float sm = 0.0f;
        for (int m = 0; m < NJ; ++m) { const float e = expf(row[m] - mx); row[m] = e; sm += e; }
        const float inv = 1.0f / sm;
        for (int m = 0; m < NJ; ++m) row[m] *= inv;
    }
    __syncthreads();
    if (tid < NJ) {
        float s = 0.0f;
        for (int hh = 0; hh < 4; ++hh)
            for (int m = 0; m < NJ; ++m) s += S[(hh * NJ + tid) * NJ + m];
        atomicAdd(&scores[b * NJ + tid], s * (1.0f / HEADS));
    }
    for (int idx = tid; idx < 4 * NJ * HDC; idx += 256) {
        const int hh = idx / (NJ * HDC), r = idx % (NJ * HDC), n = r / HDC, d = r % HDC;
        const float* srow = &S[(hh * NJ + n) * NJ];
        float acc = 0.0f;
        for (int m = 0; m < NJ; ++m) acc += srow[m] * vs[(hh * NJ + m) * HDC + d];
        ao[((size_t)b * NJ + n) * DIMC + (hg * 4 + hh) * HDC + d] = acc;
    }
}

// ---------------------------------------------------------------------------
// Globe attention: s = relu(qk * rd)/HD, softmax; go = attn@v + f (f = xn2)
// ---------------------------------------------------------------------------
__global__ __launch_bounds__(256)
void gl_attn_kernel(const float* __restrict__ qkv, const float* __restrict__ xn2,
                    const float* __restrict__ rdm, float* __restrict__ go)
{
    __shared__ float qs[4 * NJ * HDC], ks[4 * NJ * HDC], vs[4 * NJ * HDC];
    __shared__ float S[4 * NJ * NJ];
    __shared__ float Rd[NJ * NJ];
    const int b = blockIdx.x, hg = blockIdx.y, tid = threadIdx.x;
    for (int idx = tid; idx < 4 * NJ * HDC; idx += 256) {
        const int hh = idx / (NJ * HDC), r = idx % (NJ * HDC), n = r / HDC, d = r % HDC;
        const int h = hg * 4 + hh;
        const size_t base = ((size_t)b * NJ + n) * (3 * DIMC);
        qs[idx] = qkv[base + h * HDC + d];
        ks[idx] = qkv[base + DIMC + h * HDC + d];
        vs[idx] = qkv[base + 2 * DIMC + h * HDC + d];
    }
    for (int idx = tid; idx < NJ * NJ; idx += 256) Rd[idx] = rdm[idx];
    __syncthreads();
    for (int idx = tid; idx < 4 * NJ * NJ; idx += 256) {
        const int hh = idx / (NJ * NJ), r = idx % (NJ * NJ), n = r / NJ, m = r % NJ;
        const float* qp = &qs[(hh * NJ + n) * HDC];
        const float* kp = &ks[(hh * NJ + m) * HDC];
        float acc = 0.0f;
        for (int d = 0; d < HDC; ++d) acc += qp[d] * kp[d];
        acc *= Rd[n * NJ + m];
        acc = fmaxf(acc, 0.0f);
        S[idx] = acc * (1.0f / 64.0f);   // / sqrt(HD) * HD^-0.5
    }
    __syncthreads();
    for (int idx = tid; idx < 4 * NJ; idx += 256) {
        float* row = &S[idx * NJ];
        float mx = row[0];
        for (int m = 1; m < NJ; ++m) mx = fmaxf(mx, row[m]);
        float sm = 0.0f;
        for (int m = 0; m < NJ; ++m) { const float e = expf(row[m] - mx); row[m] = e; sm += e; }
        const float inv = 1.0f / sm;
        for (int m = 0; m < NJ; ++m) row[m] *= inv;
    }
    __syncthreads();
    for (int idx = tid; idx < 4 * NJ * HDC; idx += 256) {
        const int hh = idx / (NJ * HDC), r = idx % (NJ * HDC), n = r / HDC, d = r % HDC;
        const float* srow = &S[(hh * NJ + n) * NJ];
        float acc = 0.0f;
        for (int m = 0; m < NJ; ++m) acc += srow[m] * vs[(hh * NJ + m) * HDC + d];
        const size_t o = ((size_t)b * NJ + n) * DIMC + (hg * 4 + hh) * HDC + d;
        go[o] = acc + xn2[o];
    }
}

// ---------------------------------------------------------------------------
// Dynamic fusion: argmax score -> root estimate -> BFS along skeleton
// x3 = (root - pathsum) * fa + x2, in place on X
// ---------------------------------------------------------------------------
__global__ __launch_bounds__(256)
void dynfuse_kernel(float* __restrict__ X, const float* __restrict__ V,
                    const float* __restrict__ scores, const float* __restrict__ fav)
{
    __shared__ float rootv[DIMC];
    __shared__ int s_top;
    const int b = blockIdx.x, tid = threadIdx.x;
    if (tid == 0) {
        float best = scores[b * NJ];
        int bi = 0;
        for (int n = 1; n < NJ; ++n) {
            const float v = scores[b * NJ + n];
            if (v > best) { best = v; bi = n; }
        }
        s_top = bi;
    }
    __syncthreads();
    const int top1 = s_top;
    const int aj = (top1 == 0) ? 0 : cANC[top1];
    for (int c = tid; c < DIMC; c += 256) {
        float r = X[((size_t)b * NJ + aj) * DIMC + c];
        if (top1 != 0) r += V[((size_t)b * NJ + (aj - 1)) * DIMC + c];
        rootv[c] = r;
    }
    __syncthreads();
    const float fa = fav[0];
    for (int idx = tid; idx < NJ * DIMC; idx += 256) {
        const int j = idx / DIMC, c = idx % DIMC;
        float ps = 0.0f;
        int jj = j;
        while (jj != 0) { ps += V[((size_t)b * NJ + (jj - 1)) * DIMC + c]; jj = cPAR[jj]; }
        const size_t o = ((size_t)b * NJ + j) * DIMC + c;
        X[o] = (rootv[c] - ps) * fa + X[o];
    }
}

// ---------------------------------------------------------------------------
extern "C" void kernel_launch(void* const* d_in, const int* in_sizes, int n_in,
                              void* d_out, int out_size, void* d_ws, size_t ws_size,
                              hipStream_t stream)
{
    const float* x      = (const float*)d_in[0];
    const float* vector = (const float*)d_in[1];
    const float* ew     = (const float*)d_in[2];
    const float* rdist  = (const float*)d_in[3];
    const float* adjk   = (const float*)d_in[4];
    const float* dadj   = (const float*)d_in[5];
    const float* jgcn_w = (const float*)d_in[6];
    const float* jgcn_b = (const float*)d_in[7];
    const float* gcn1_w = (const float*)d_in[8];
    const float* gcn1_b = (const float*)d_in[9];
    const float* bc2_w  = (const float*)d_in[10];
    const float* bc2_b  = (const float*)d_in[11];
    const float* bc4_w  = (const float*)d_in[12];
    const float* bc4_b  = (const float*)d_in[13];
    const float* co_qkv_w = (const float*)d_in[14];
    const float* co_qkv_b = (const float*)d_in[15];
    const float* co_pw  = (const float*)d_in[16];
    const float* co_pb  = (const float*)d_in[17];
    const float* gl_qkv_w = (const float*)d_in[18];
    const float* gl_qkv_b = (const float*)d_in[19];
    const float* gl_pw  = (const float*)d_in[20];
    const float* gl_pb  = (const float*)d_in[21];
    const float* ffn_w1 = (const float*)d_in[22];
    const float* ffn_b1 = (const float*)d_in[23];
    const float* ffn_w2 = (const float*)d_in[24];
    const float* ffn_b2 = (const float*)d_in[25];
    const float* ffn_g  = (const float*)d_in[26];
    const float* ffn_be = (const float*)d_in[27];
    const float* natt_g = (const float*)d_in[28];
    const float* natt_b = (const float*)d_in[29];
    const float* n2_g   = (const float*)d_in[30];
    const float* n2_b   = (const float*)d_in[31];
    const float* fa     = (const float*)d_in[32];
    const int*   eidx   = (const int*)d_in[33];

    // workspace layout (floats); lifetime-based aliasing
    float* w = (float*)d_ws;
    const size_t NX = (size_t)MROWS * DIMC;       // 17,825,792
    float* Q  = w;                                 // 3*NX: H / T1,T3,T2 / QKV / QKV2 / G
    float* X  = w + 3 * NX;                        // x1 -> x2 -> x3 -> x4 -> ffn-out (in place)
    float* V  = w + 4 * NX;                        // vec, later go
    float* Nb = w + 5 * NX;                        // xn / ao / xn2
    float* SC = w + 6 * NX;                        // scores (MROWS)
    float* AH = SC + MROWS;                        // Ahat (289)
    float* RD = AH + NJ * NJ;                      // rd   (289)
    float* T1 = Q;
    float* T3 = Q + NX;
    float* T2 = Q + 2 * NX;

    const dim3 blk(128);
    auto g2 = [](int nd) { return dim3(MROWS / 32, nd / 256); };
    (void)in_sizes; (void)n_in; (void)out_size; (void)ws_size;

    // 0) graph constants + zero scores
    prep_kernel<<<1, 64, 0, stream>>>(ew, rdist, eidx, AH, RD);
    zero_kernel<<<(MROWS + 255) / 256, 256, 0, stream>>>(SC, MROWS);

    // 1) Joint_GCN: H = x @ jgcn_w^T + b ; x1 = x + einsum(H, adj_k)
    gemm_wmma_kernel<<<g2(1536), blk, 0, stream>>>(x, jgcn_w, jgcn_b, nullptr, Q, 512, 1536, 0);
    jgcn_mix_kernel<<<dim3(BATCH, 2), 256, 0, stream>>>(x, Q, adjk, X);

    // 2) Bone branch: T1 = vector@gcn1_w^T ; T2 = leaky(vector@bc2^T+b) ; T3 = T2@bc4^T+b
    gemm_wmma_kernel<<<g2(512), blk, 0, stream>>>(vector, gcn1_w, nullptr, nullptr, T1, 512, 512, 0);
    gemm_wmma_kernel<<<g2(256), blk, 0, stream>>>(vector, bc2_w, bc2_b, nullptr, T2, 512, 256, 1);
    gemm_wmma_kernel<<<g2(512), blk, 0, stream>>>(T2, bc4_w, bc4_b, nullptr, T3, 256, 512, 0);
    bone_mix_kernel<<<dim3(BATCH, 2), 256, 0, stream>>>(vector, T1, T3, AH, dadj, gcn1_b, V);

    // 3) CoAttention
    ln_kernel<<<MROWS / 8, 256, 0, stream>>>(X, natt_g, natt_b, Nb);
    gemm_wmma_kernel<<<g2(1536), blk, 0, stream>>>(Nb, co_qkv_w, co_qkv_b, nullptr, Q, 512, 1536, 0);
    co_attn_kernel<<<dim3(BATCH, 2), 256, 0, stream>>>(Q, Nb, SC);      // ao overwrites xn
    gemm_wmma_kernel<<<g2(512), blk, 0, stream>>>(Nb, co_pw, co_pb, X, X, 512, 512, 0);

    // 4) Dynamic fusion (in place on X, consumes vec + scores)
    dynfuse_kernel<<<BATCH, 256, 0, stream>>>(X, V, SC, fa);

    // 5) Globe attention
    ln_kernel<<<MROWS / 8, 256, 0, stream>>>(X, n2_g, n2_b, Nb);
    gemm_wmma_kernel<<<g2(1536), blk, 0, stream>>>(Nb, gl_qkv_w, gl_qkv_b, nullptr, Q, 512, 1536, 0);
    gl_attn_kernel<<<dim3(BATCH, 2), 256, 0, stream>>>(Q, Nb, RD, V);   // go overwrites vec
    gemm_wmma_kernel<<<g2(512), blk, 0, stream>>>(V, gl_pw, gl_pb, X, X, 512, 512, 0);

    // 6) FFN + post-norm
    gemm_wmma_kernel<<<g2(256), blk, 0, stream>>>(X, ffn_w1, ffn_b1, nullptr, Q, 512, 256, 2);
    gemm_wmma_kernel<<<g2(512), blk, 0, stream>>>(Q, ffn_w2, ffn_b2, X, X, 256, 512, 0);
    ln_kernel<<<MROWS / 8, 256, 0, stream>>>(X, ffn_g, ffn_be, (float*)d_out);
}
